// HFMoEBlock_44959717655037
// MI455X (gfx1250) — matrix operand order repaired and checked by reading, not compiled, not run on hardware
//
#include <hip/hip_runtime.h>
#include <hip/hip_bf16.h>
#include <math.h>

// ---------------- problem constants ----------------
#define T_TOKENS 8192
#define H_DIM    2048
#define F_DIM    5632
#define N_EXPERTS 8

// ---------------- tiling ----------------
#define M_TILE 32            // tokens per block (2 WMMA M-subtiles)
#define FT     64            // F tile width (4 WMMA N-subtiles per matrix)
#define XSTR   (H_DIM + 8)   // 2056 bf16 elems; rows land 4 banks apart -> conflict free
#define GUSTR  68            // f32 stride for g/u staging
#define HSTR   88            // bf16 stride for h staging (176B, 16B aligned)

typedef __attribute__((ext_vector_type(16))) __bf16       v16bf;
typedef __attribute__((ext_vector_type(8)))  float        v8f;
typedef __attribute__((ext_vector_type(4)))  unsigned int u32x4;
typedef __attribute__((ext_vector_type(2)))  unsigned int u32x2;
typedef __attribute__((ext_vector_type(4)))  float        f32x4;
typedef __attribute__((ext_vector_type(2)))  float        f32x2;
typedef __attribute__((ext_vector_type(2)))  __bf16       bf16x2;

union ABu { u32x4 q[2]; v16bf v; };

// fp32 pair -> packed bf16 (prefer single v_cvt_pk_bf16_f32)
#if __has_builtin(__builtin_amdgcn_cvt_pk_bf16_f32)
__device__ __forceinline__ unsigned pk_bf16(float a, float b) {
  bf16x2 r = __builtin_amdgcn_cvt_pk_bf16_f32(a, b);
  return __builtin_bit_cast(unsigned, r);
}
#else
__device__ __forceinline__ unsigned pk_bf16(float a, float b) {
  unsigned ua = __float_as_uint(a);
  unsigned ub = __float_as_uint(b);
  ua += 0x7fffu + ((ua >> 16) & 1u);
  ub += 0x7fffu + ((ub >> 16) & 1u);
  return (ua >> 16) | (ub & 0xffff0000u);
}
#endif

#if __has_builtin(__builtin_amdgcn_rcpf)
__device__ __forceinline__ float fast_rcp(float a) { return __builtin_amdgcn_rcpf(a); }
#else
__device__ __forceinline__ float fast_rcp(float a) { return 1.f / a; }
#endif

// A operand (16 x K=32 bf16) from an LDS row-major tile.
// Lane L (row = L&15): 16 consecutive K at k0 + (L<16 ? 0 : 16).
__device__ __forceinline__ v16bf load_a_lds(const unsigned short* tile, int stride,
                                            int k0, int lane) {
  const int row = lane & 15;
  const int col = k0 + ((lane & 16) ? 16 : 0);
  const unsigned short* p = tile + row * stride + col;
  ABu u;
  u.q[0] = *(const u32x4*)p;
  u.q[1] = *(const u32x4*)(p + 8);
  return u.v;
}

// B operand (K=32 x 16): lane L holds column n = L&15 = one weight row,
// 16 consecutive K values at k0 + (L<16 ? 0 : 16).
__device__ __forceinline__ v16bf load_b_f32(const float* rowbase, int ld,
                                            int k0, int lane) {
  const float* p = rowbase + (size_t)(lane & 15) * (size_t)ld + k0 + ((lane & 16) ? 16 : 0);
  f32x4 a = *(const f32x4*)(p);
  f32x4 b = *(const f32x4*)(p + 4);
  f32x4 c = *(const f32x4*)(p + 8);
  f32x4 d = *(const f32x4*)(p + 12);
  ABu u;
  u.q[0] = (u32x4){ pk_bf16(a.x, a.y), pk_bf16(a.z, a.w),
                    pk_bf16(b.x, b.y), pk_bf16(b.z, b.w) };
  u.q[1] = (u32x4){ pk_bf16(c.x, c.y), pk_bf16(c.z, c.w),
                    pk_bf16(d.x, d.y), pk_bf16(d.z, d.w) };
  return u.v;
}

__device__ __forceinline__ v16bf load_b_bf16(const unsigned short* rowbase, int ld,
                                             int k0, int lane) {
  const unsigned short* p = rowbase + (size_t)(lane & 15) * (size_t)ld + k0 +
                            ((lane & 16) ? 16 : 0);
  ABu u;
  u.q[0] = *(const u32x4*)p;        // 8 bf16
  u.q[1] = *(const u32x4*)(p + 8);  // 8 bf16
  return u.v;
}

__device__ __forceinline__ v8f wmma_bf16(v16bf a, v16bf b, v8f c) {
  return __builtin_amdgcn_wmma_f32_16x16x32_bf16(false, a, false, b, (short)0, c,
                                                 false, false);
}

// ---------------- kernel 0: zero output + expert counters ----------------
__global__ void zero_kernel(float* __restrict__ out, int* __restrict__ cnt, long n4) {
  long gid = (long)blockIdx.x * blockDim.x + threadIdx.x;
  if (gid < n4) {
    f32x4 z = {0.f, 0.f, 0.f, 0.f};
    *(f32x4*)(out + gid * 4) = z;
  }
  if (gid < N_EXPERTS) cnt[gid] = 0;
}

// ---------------- kernel 0b: weights f32 -> bf16 (one pass per call) ----------------
__global__ void convert_w_kernel(const float* __restrict__ src,
                                 unsigned int* __restrict__ dst, long n4) {
  long gid = (long)blockIdx.x * blockDim.x + threadIdx.x;
  if (gid < n4) {
    f32x4 v = *(const f32x4*)(src + gid * 4);
    u32x2 o = { pk_bf16(v.x, v.y), pk_bf16(v.z, v.w) };
    *(u32x2*)(dst + gid * 2) = o;
  }
}

// ---------------- kernel 1: router (one wave per token) ----------------
__global__ __launch_bounds__(256)
void router_kernel(const float* __restrict__ x, const float* __restrict__ gw,
                   float* __restrict__ rlog, int* __restrict__ cnt,
                   int* __restrict__ idxl, float* __restrict__ wl) {
  const int lane = threadIdx.x & 31;
  const int wv   = threadIdx.x >> 5;
  const int t    = blockIdx.x * 8 + wv;

  float acc[N_EXPERTS];
#pragma unroll
  for (int e = 0; e < N_EXPERTS; ++e) acc[e] = 0.f;

  const float* xr = x + (size_t)t * H_DIM;
  for (int c = lane * 4; c < H_DIM; c += 128) {
    f32x4 xv = *(const f32x4*)(xr + c);
#pragma unroll
    for (int e = 0; e < N_EXPERTS; ++e) {
      f32x4 gv = *(const f32x4*)(gw + (size_t)e * H_DIM + c);
      acc[e] += xv.x * gv.x + xv.y * gv.y + xv.z * gv.z + xv.w * gv.w;
    }
  }
#pragma unroll
  for (int e = 0; e < N_EXPERTS; ++e) {
    float v = acc[e];
#pragma unroll
    for (int off = 16; off >= 1; off >>= 1) v += __shfl_xor(v, off, 32);
    acc[e] = v;
  }

  if (lane < N_EXPERTS) rlog[(size_t)t * N_EXPERTS + lane] = acc[lane];

  // top-2 (earliest index wins ties, matching lax.top_k)
  int i1 = 0; float m1 = acc[0];
#pragma unroll
  for (int e = 1; e < N_EXPERTS; ++e) if (acc[e] > m1) { m1 = acc[e]; i1 = e; }
  int i2 = -1; float m2 = -3.402823e38f;
#pragma unroll
  for (int e = 0; e < N_EXPERTS; ++e)
    if (e != i1 && acc[e] > m2) { m2 = acc[e]; i2 = e; }

  if (lane == 0) {
    float b  = __expf(m2 - m1);   // renormalized softmax over winners
    float wa = 1.f / (1.f + b);
    float wb = b / (1.f + b);
    int p1 = atomicAdd(&cnt[i1], 1);
    idxl[i1 * T_TOKENS + p1] = t;  wl[i1 * T_TOKENS + p1] = wa;
    int p2 = atomicAdd(&cnt[i2], 1);
    idxl[i2 * T_TOKENS + p2] = t;  wl[i2 * T_TOKENS + p2] = wb;
  }
}

// ---------------- kernel 2: fused expert SwiGLU MLP ----------------
// block = (expert, 32-token tile), 512 threads = 16 wave32.
// BF16W: weights already converted to bf16 in workspace (pure b128 B loads).
template <bool BF16W>
__global__ __launch_bounds__(512, 1)
void moe_expert_kernel(const float* __restrict__ x,
                       const void*  __restrict__ w1v,
                       const void*  __restrict__ w3v,
                       const void*  __restrict__ w2v,
                       const int*   __restrict__ cnt,
                       const int*   __restrict__ idxl,
                       const float* __restrict__ wl,
                       float* __restrict__ out) {
  __shared__ unsigned short xt[M_TILE * XSTR];       // x tile, bf16   (~128.5 KB)
  __shared__ float          gu[2][M_TILE][GUSTR];    // g/u staging    (~17 KB)
  __shared__ unsigned short hls[M_TILE * HSTR];      // h tile, bf16   (~5.5 KB)
  __shared__ int            tok_s[M_TILE];
  __shared__ float          cw_s[M_TILE];

  const int e    = blockIdx.y;
  const int base = blockIdx.x * M_TILE;
  const int ce   = cnt[e];
  if (base >= ce) return;                 // uniform: EXEC all-ones for WMMA
  const int mv = min(M_TILE, ce - base);

  const int tid  = threadIdx.x;
  const int lane = tid & 31;
  const int wv   = tid >> 5;

  if (tid < M_TILE) {
    if (tid < mv) {
      tok_s[tid] = idxl[e * T_TOKENS + base + tid];
      cw_s[tid]  = wl[e * T_TOKENS + base + tid];
    } else { tok_s[tid] = 0; cw_s[tid] = 0.f; }
  }
  __syncthreads();

  // zero the x tile (padded rows must be 0), then gather valid token rows
  unsigned int* xtw = (unsigned int*)xt;
  for (int i = tid; i < M_TILE * (XSTR / 2); i += 512) xtw[i] = 0u;
  __syncthreads();
  for (int p = tid; p < M_TILE * (H_DIM / 2); p += 512) {
    int m  = p >> 10;                    // H_DIM/2 = 1024 pairs per row
    int c2 = p & 1023;
    if (m < mv) {
      const float* xp = x + (size_t)tok_s[m] * H_DIM + c2 * 2;
      f32x2 v = *(const f32x2*)xp;
      xtw[m * (XSTR / 2) + c2] = pk_bf16(v.x, v.y);
    }
  }
  __syncthreads();

  // GEMM1 wave roles: 2 matrices x 2 M-subtiles x 4 N-subtiles = 16 waves
  const int mat   = wv & 1;
  const int msub1 = (wv >> 1) & 1;
  const int nsub1 = (wv >> 2) & 3;
  const int col0  = wv * 128;            // GEMM2: 128-column strip of H per wave

  const v8f vzero = {0.f, 0.f, 0.f, 0.f, 0.f, 0.f, 0.f, 0.f};
  v8f acc[2][8];                          // 32 x 128 f32 strip in VGPRs
#pragma unroll
  for (int i = 0; i < 2; ++i)
#pragma unroll
    for (int j = 0; j < 8; ++j) acc[i][j] = vzero;

  for (int f0 = 0; f0 < F_DIM; f0 += FT) {
    // ---- phase 1: this wave's 16x16 subtile of g or u, K-loop over H ----
    // Software-pipelined (double-buffered operands, last iteration peeled).
    v8f cg = vzero;
    const size_t brow1 = ((size_t)e * F_DIM + f0 + nsub1 * 16) * H_DIM;
    const unsigned short* bb16 =
        (BF16W ? ((const unsigned short*)(mat ? w3v : w1v) + brow1) : (const unsigned short*)0);
    const float* bb32 =
        (!BF16W ? ((const float*)(mat ? w3v : w1v) + brow1) : (const float*)0);
    const unsigned short* abase1 = xt + msub1 * 16 * XSTR;

    // warm WGP$/L2 for the next f-tile's weight rows (global_prefetch_b8)
    if (f0 + FT < F_DIM) {
      if (BF16W) __builtin_prefetch(bb16 + (size_t)FT * H_DIM, 0, 0);
      else       __builtin_prefetch(bb32 + (size_t)FT * H_DIM, 0, 0);
    }

    v16bf a_cur = load_a_lds(abase1, XSTR, 0, lane);
    v16bf b_cur = BF16W ? load_b_bf16(bb16, H_DIM, 0, lane)
                        : load_b_f32 (bb32, H_DIM, 0, lane);
    for (int k0 = 0; k0 < H_DIM - 32; k0 += 32) {
      v16bf a_nxt = load_a_lds(abase1, XSTR, k0 + 32, lane);
      v16bf b_nxt = BF16W ? load_b_bf16(bb16, H_DIM, k0 + 32, lane)
                          : load_b_f32 (bb32, H_DIM, k0 + 32, lane);
      cg = wmma_bf16(a_cur, b_cur, cg);
      a_cur = a_nxt;
      b_cur = b_nxt;
    }
    cg = wmma_bf16(a_cur, b_cur, cg);   // peeled last iteration

    {
      const int mrow = msub1 * 16 + ((lane & 16) ? 8 : 0);
      const int ncol = nsub1 * 16 + (lane & 15);
#pragma unroll
      for (int r = 0; r < 8; ++r) gu[mat][mrow + r][ncol] = cg[r];
    }
    __syncthreads();

    // ---- SwiGLU: h = silu(g) * u = g * rcp(1+exp(-g)) * u, store bf16 ----
    for (int p = tid; p < M_TILE * FT; p += 512) {
      int m = p >> 6, n = p & 63;
      float g  = gu[0][m][n];
      float uu = gu[1][m][n];
      float hv = g * fast_rcp(1.f + __expf(-g)) * uu;
      hls[m * HSTR + n] = (unsigned short)(pk_bf16(hv, 0.f) & 0xffffu);
    }
    __syncthreads();

    // ---- phase 2: out strip += h(32x64) @ w2[col0..col0+127, f0..f0+63]^T ----
    // B operands for subtile s+1 are fetched while subtile s's WMMAs issue.
    v16bf a0[2], a1[2];
#pragma unroll
    for (int ms = 0; ms < 2; ++ms) {
      a0[ms] = load_a_lds(hls + ms * 16 * HSTR, HSTR, 0,  lane);
      a1[ms] = load_a_lds(hls + ms * 16 * HSTR, HSTR, 32, lane);
    }
    const size_t brow2base = ((size_t)e * H_DIM + col0) * F_DIM + f0;
    v16bf b0c, b1c;
    if (BF16W) {
      const unsigned short* bb = (const unsigned short*)w2v + brow2base;
      b0c = load_b_bf16(bb, F_DIM, 0,  lane);
      b1c = load_b_bf16(bb, F_DIM, 32, lane);
    } else {
      const float* bb = (const float*)w2v + brow2base;
      b0c = load_b_f32(bb, F_DIM, 0,  lane);
      b1c = load_b_f32(bb, F_DIM, 32, lane);
    }
#pragma unroll
    for (int s = 0; s < 8; ++s) {
      v16bf b0n, b1n;
      if (s < 7) {
        const size_t brow2 = brow2base + (size_t)(s + 1) * 16 * F_DIM;
        if (BF16W) {
          const unsigned short* bb = (const unsigned short*)w2v + brow2;
          b0n = load_b_bf16(bb, F_DIM, 0,  lane);
          b1n = load_b_bf16(bb, F_DIM, 32, lane);
        } else {
          const float* bb = (const float*)w2v + brow2;
          b0n = load_b_f32(bb, F_DIM, 0,  lane);
          b1n = load_b_f32(bb, F_DIM, 32, lane);
        }
      }
      acc[0][s] = wmma_bf16(a0[0], b0c, acc[0][s]);
      acc[1][s] = wmma_bf16(a0[1], b0c, acc[1][s]);
      acc[0][s] = wmma_bf16(a1[0], b1c, acc[0][s]);
      acc[1][s] = wmma_bf16(a1[1], b1c, acc[1][s]);
      if (s < 7) { b0c = b0n; b1c = b1n; }
    }
    __syncthreads();   // gu/hls reused next f-tile
  }

  // ---- scatter: out[token, :] += cw * strip (exactly 2 adds/elem -> commutative) ----
  const int hi8 = (lane & 16) ? 8 : 0;
  const int nl  = lane & 15;
#pragma unroll
  for (int ms = 0; ms < 2; ++ms) {
#pragma unroll
    for (int r = 0; r < 8; ++r) {
      const int m = ms * 16 + r + hi8;
      if (m < mv) {
        const float cw = cw_s[m];
        float* orow = out + (size_t)tok_s[m] * H_DIM + col0 + nl;
#pragma unroll
        for (int s = 0; s < 8; ++s) atomicAdd(orow + s * 16, cw * acc[ms][s][r]);
      }
    }
  }
}

// ---------------- host ----------------
extern "C" void kernel_launch(void* const* d_in, const int* in_sizes, int n_in,
                              void* d_out, int out_size, void* d_ws, size_t ws_size,
                              hipStream_t stream) {
  const float* x  = (const float*)d_in[0];   // [T, H]
  const float* gw = (const float*)d_in[1];   // [E, H]
  const float* w1 = (const float*)d_in[2];   // [E, F, H]
  const float* w3 = (const float*)d_in[3];   // [E, F, H]
  const float* w2 = (const float*)d_in[4];   // [E, H, F]

  float* out  = (float*)d_out;                       // [T, H]
  float* rlog = out + (size_t)T_TOKENS * H_DIM;      // [T, E]

  char*  wsb  = (char*)d_ws;
  int*   cnt  = (int*)wsb;                                                 // [E]
  int*   idxl = (int*)(wsb + 64);                                          // [E, T]
  float* wl   = (float*)(wsb + 64 + sizeof(int) * N_EXPERTS * T_TOKENS);   // [E, T]

  const size_t LIST_RESERVE = 1u << 20;                    // 1 MB for lists/counters
  const size_t WELEMS = (size_t)N_EXPERTS * F_DIM * H_DIM; // per weight tensor
  const bool bf16path = ws_size >= LIST_RESERVE + 3 * WELEMS * sizeof(unsigned short);

  const long n4 = (long)T_TOKENS * H_DIM / 4;
  zero_kernel<<<(unsigned)((n4 + 255) / 256), 256, 0, stream>>>(out, cnt, n4);
  router_kernel<<<T_TOKENS / 8, 256, 0, stream>>>(x, gw, rlog, cnt, idxl, wl);

  dim3 grid((T_TOKENS + M_TILE - 1) / M_TILE, N_EXPERTS);
  if (bf16path) {
    unsigned short* w1b = (unsigned short*)(wsb + LIST_RESERVE);
    unsigned short* w3b = w1b + WELEMS;
    unsigned short* w2b = w3b + WELEMS;
    const long wn4 = (long)WELEMS / 4;
    const unsigned cblk = (unsigned)((wn4 + 255) / 256);
    convert_w_kernel<<<cblk, 256, 0, stream>>>(w1, (unsigned int*)w1b, wn4);
    convert_w_kernel<<<cblk, 256, 0, stream>>>(w3, (unsigned int*)w3b, wn4);
    convert_w_kernel<<<cblk, 256, 0, stream>>>(w2, (unsigned int*)w2b, wn4);
    moe_expert_kernel<true><<<grid, 512, 0, stream>>>(x, w1b, w3b, w2b,
                                                      cnt, idxl, wl, out);
  } else {
    moe_expert_kernel<false><<<grid, 512, 0, stream>>>(x, w1, w3, w2,
                                                       cnt, idxl, wl, out);
  }
}
